// NodeAttention_23871428231777
// MI455X (gfx1250) — matrix operand order repaired and checked by reading, not compiled, not run on hardware
//
#include <hip/hip_runtime.h>
#include <hip/hip_bf16.h>
#include <math.h>

// ---------------------------------------------------------------------------
// Problem constants (B=8, N=512, H=16, D=512, DH=32)
// ---------------------------------------------------------------------------
#define BB 8
#define NN 512
#define HH 16
#define DD 512
#define DHH 32
#define ROWS (BB * NN)   // 4096
#define LN_EPS 1e-5f
#define TS 36            // LDS tile row stride (floats): conflict-free & 16B-aligned

typedef __bf16 v16bf __attribute__((ext_vector_type(16)));
typedef float  v8f   __attribute__((ext_vector_type(8)));

// split a f32 into bf16 hi + bf16 lo correction (3xBF16 fp32-emulation)
static __device__ __forceinline__ void splitpush(v16bf& hi, v16bf& lo, int e, float x) {
    __bf16 h = (__bf16)x;
    hi[e] = h;
    lo[e] = (__bf16)(x - (float)h);
}

// D += A*B with 3-term bf16 split: ah*bh + al*bh + ah*bl  (~fp32 accuracy)
static __device__ __forceinline__ v8f wmma3(const v16bf& ah, const v16bf& al,
                                            const v16bf& bh, const v16bf& bl, v8f acc) {
    acc = __builtin_amdgcn_wmma_f32_16x16x32_bf16(false, ah, false, bh, (short)0, acc, false, false);
    acc = __builtin_amdgcn_wmma_f32_16x16x32_bf16(false, al, false, bh, (short)0, acc, false, false);
    acc = __builtin_amdgcn_wmma_f32_16x16x32_bf16(false, ah, false, bl, (short)0, acc, false, false);
    return acc;
}

// async 16B copy global -> LDS, tracked with ASYNCcnt (per-lane addresses)
static __device__ __forceinline__ void async_b128(unsigned lds_off, const float* g) {
    asm volatile("global_load_async_to_lds_b128 %0, %1, off"
                 :: "v"(lds_off), "v"(g) : "memory");
}
static __device__ __forceinline__ unsigned lds_off_of(const float* p) {
    return (unsigned)(size_t)p;   // generic shared ptr low 32 bits = LDS byte addr
}

// ---------------------------------------------------------------------------
// Kernel 1: LayerNorm.  One wave32 per row of 512 elements.
// ---------------------------------------------------------------------------
__global__ void ln_kernel(const float* __restrict__ x, const float* __restrict__ g,
                          const float* __restrict__ be, float* __restrict__ y) {
    const int row  = blockIdx.x * 8 + (threadIdx.x >> 5);
    const int lane = threadIdx.x & 31;
    const float* xr = x + (size_t)row * DD;
    float v[16];
    float s = 0.f, s2 = 0.f;
#pragma unroll
    for (int k = 0; k < 16; ++k) {
        float t = xr[lane + 32 * k];
        v[k] = t; s += t; s2 += t * t;
    }
#pragma unroll
    for (int off = 16; off; off >>= 1) {
        s  += __shfl_xor(s,  off);
        s2 += __shfl_xor(s2, off);
    }
    const float mean = s * (1.f / DD);
    const float var  = s2 * (1.f / DD) - mean * mean;
    const float rstd = rsqrtf(var + LN_EPS);
    float* yr = y + (size_t)row * DD;
#pragma unroll
    for (int k = 0; k < 16; ++k) {
        const int c = lane + 32 * k;
        yr[c] = (v[k] - mean) * rstd * g[c] + be[c];
    }
}

// ---------------------------------------------------------------------------
// Kernel 2: C[M,512] = A[M,512] @ W[512,512]^T + bias, via bf16x3 WMMA.
// Block = 128 threads (4 waves); wave computes a 16(M) x 32(N) tile; all 4
// waves share one A tile, which is double-buffered through LDS with async
// global->LDS copies (ASYNCcnt).  grid = (M/16, 512/128).
// ---------------------------------------------------------------------------
__global__ void gemm_bias_kernel(const float* __restrict__ A, const float* __restrict__ W,
                                 const float* __restrict__ bias, float* __restrict__ C) {
    __shared__ float tileA[2][16 * TS];

    const int tid  = threadIdx.x;
    const int wave = tid >> 5;
    const int lane = tid & 31;
    const int row0 = blockIdx.x * 16;
    const int n0   = blockIdx.y * 128 + wave * 32;
    const int r    = lane & 15;
    const int half = lane >> 4;

    // this thread's async chunk: row = tid>>3 (0..15), 16B segment = tid&7
    const int srow = tid >> 3, sseg = tid & 7;
    const float* Asrc = A + (size_t)(row0 + srow) * DD + sseg * 4;

    v8f acc0 = {}, acc1 = {};
    const float* Wr0 = W + (size_t)(n0 + r) * DD;        // B col n -> W row (n0+n)
    const float* Wr1 = W + (size_t)(n0 + 16 + r) * DD;

    // prologue: stage k-chunk 0
    async_b128(lds_off_of(&tileA[0][srow * TS + sseg * 4]), Asrc);

    for (int kk = 0; kk < 16; ++kk) {
        const int k0 = kk * 32;
        if (kk < 15) {
            async_b128(lds_off_of(&tileA[(kk + 1) & 1][srow * TS + sseg * 4]),
                       Asrc + (k0 + 32));
            asm volatile("s_wait_asynccnt 0x1" ::: "memory");
        } else {
            asm volatile("s_wait_asynccnt 0x0" ::: "memory");
        }
        __syncthreads();

        const float* ta = &tileA[kk & 1][0];
        v16bf ah, al;
#pragma unroll
        for (int p = 0; p < 8; ++p) {
            const int kb = half * 16 + 2 * p;
            const float2 va = *(const float2*)(ta + r * TS + kb);
            splitpush(ah, al, 2 * p,     va.x);
            splitpush(ah, al, 2 * p + 1, va.y);
        }
        // B sub-tile 0, then sub-tile 1 (reuse fragment regs to limit pressure)
        {
            v16bf bh, bl;
            const int kb = k0 + half * 16;
#pragma unroll
            for (int q = 0; q < 4; ++q) {
                const float4 w0 = *(const float4*)(Wr0 + kb + 4 * q);
                splitpush(bh, bl, 4 * q + 0, w0.x); splitpush(bh, bl, 4 * q + 1, w0.y);
                splitpush(bh, bl, 4 * q + 2, w0.z); splitpush(bh, bl, 4 * q + 3, w0.w);
            }
            acc0 = wmma3(ah, al, bh, bl, acc0);
        }
        {
            v16bf bh, bl;
            const int kb = k0 + half * 16;
#pragma unroll
            for (int q = 0; q < 4; ++q) {
                const float4 w1 = *(const float4*)(Wr1 + kb + 4 * q);
                splitpush(bh, bl, 4 * q + 0, w1.x); splitpush(bh, bl, 4 * q + 1, w1.y);
                splitpush(bh, bl, 4 * q + 2, w1.z); splitpush(bh, bl, 4 * q + 3, w1.w);
            }
            acc1 = wmma3(ah, al, bh, bl, acc1);
        }
        __syncthreads();
    }

    const float b0 = bias[n0 + r];
    const float b1 = bias[n0 + 16 + r];
#pragma unroll
    for (int k = 0; k < 8; ++k) {           // C/D layout: row = half*8 + k, col = lane&15
        const int row = row0 + half * 8 + k;
        C[(size_t)row * DD + n0 + r]      = acc0[k] + b0;
        C[(size_t)row * DD + n0 + 16 + r] = acc1[k] + b1;
    }
}

// ---------------------------------------------------------------------------
// Kernel 3: softmax stats per (b,i,h): masked max M, then scale = sqrt(S2)/S1^2
// (folds both the softmax 1/S1 and the attn L2-norm rescale sqrt(S2)/S1).
// One 256-thread block per (b,i); thread = (s = tid>>4 j-slice, h = tid&15).
// ---------------------------------------------------------------------------
__global__ void stats_kernel(const float* __restrict__ messages, const int* __restrict__ adj,
                             float* __restrict__ Mrow, float* __restrict__ Srow,
                             int* __restrict__ deg0) {
    const int bi  = blockIdx.x;
    const int tid = threadIdx.x;
    const int h   = tid & 15;
    const int s   = tid >> 4;
    const float* mrow = messages + (size_t)bi * NN * HH;
    const int*   arow = adj + (size_t)bi * NN;

    __shared__ int   degS;
    __shared__ float red[16][17];
    __shared__ float Msh[16];
    __shared__ float S1sh[16];

    if (tid == 0) degS = 0;
    __syncthreads();
    if (h == 0) {
        int d = 0;
        for (int j = s * 32; j < s * 32 + 32; ++j) d += (arow[j] != 0);
        atomicAdd(&degS, d);
    }
    __syncthreads();
    const bool d0 = (degS == 0);   // zero-degree row: mask becomes all-zero

    float mx = -3.0e38f;
    for (int j = s * 32; j < s * 32 + 32; ++j) {
        if (d0 || arow[j]) mx = fmaxf(mx, mrow[(size_t)j * HH + h]);
    }
    red[s][h] = mx;
    __syncthreads();
    if (tid < 16) {
        float m2 = -3.0e38f;
        for (int q = 0; q < 16; ++q) m2 = fmaxf(m2, red[q][tid]);
        Msh[tid] = m2;
    }
    __syncthreads();

    const float Mh = Msh[h];
    float s1 = 0.f, s2 = 0.f;
    for (int j = s * 32; j < s * 32 + 32; ++j) {
        if (d0 || arow[j]) {
            const float e = __expf(mrow[(size_t)j * HH + h] - Mh);
            s1 += e; s2 += e * e;
        }
    }
    red[s][h] = s1;
    __syncthreads();
    if (tid < 16) { float t = 0.f; for (int q = 0; q < 16; ++q) t += red[q][tid]; S1sh[tid] = t; }
    __syncthreads();
    red[s][h] = s2;
    __syncthreads();
    if (tid < 16) {
        float S2 = 0.f; for (int q = 0; q < 16; ++q) S2 += red[q][tid];
        const float S1 = S1sh[tid];
        Mrow[(size_t)bi * HH + tid] = Msh[tid];
        Srow[(size_t)bi * HH + tid] = sqrtf(S2) / (S1 * S1);
    }
    if (tid == 0) deg0[bi] = d0 ? 1 : 0;
}

// ---------------------------------------------------------------------------
// Kernel 4: per-(b,h) aggregation out = diag(scale) * exp(m-M) * V via WMMA.
// Block = 128 threads (4 waves, same (b,h), 4 i-tiles); the shared 32(j)x32(n)
// proj tile is double-buffered through LDS with async global->LDS copies.
// grid = B * H * (N/64) = 1024.
// ---------------------------------------------------------------------------
__global__ void agg_kernel(const float* __restrict__ messages, const int* __restrict__ adj,
                           const float* __restrict__ proj, const float* __restrict__ Mrow,
                           const float* __restrict__ Srow, const int* __restrict__ deg0,
                           float* __restrict__ agg) {
    __shared__ float tileB[2][32 * TS];

    const int bid = blockIdx.x;
    const int b   = bid >> 7;           // / (H * N/64)
    const int rem = bid & 127;
    const int h   = rem >> 3;
    const int it  = rem & 7;
    const int tid  = threadIdx.x;
    const int wave = tid >> 5;
    const int lane = tid & 31;
    const int i0   = it * 64 + wave * 16;
    const int r    = lane & 15;
    const int half = lane >> 4;

    const int iA = i0 + r;                       // A-fragment row owned by this lane
    const size_t biA = (size_t)(b * NN + iA);
    const float* mrow = messages + biA * NN * HH + h;    // + j*HH
    const int*   arow = adj + biA * NN;
    const float  Mi   = Mrow[biA * HH + h];
    const bool   d0   = deg0[b * NN + iA] != 0;
    const float* projT = proj + (size_t)b * NN * DD + (size_t)h * DHH;  // + j*DD + n

    v8f acc0 = {}, acc1 = {};

    // prologue: stage j-chunk 0 (32 rows x 128B = 256 chunks; 2 per thread)
#pragma unroll
    for (int c = tid; c < 256; c += 128) {
        const int jr = c >> 3, seg = c & 7;
        async_b128(lds_off_of(&tileB[0][jr * TS + seg * 4]),
                   projT + (size_t)jr * DD + seg * 4);
    }

    for (int kk = 0; kk < 16; ++kk) {
        const int j0 = kk * 32;
        if (kk < 15) {
#pragma unroll
            for (int c = tid; c < 256; c += 128) {
                const int jr = c >> 3, seg = c & 7;
                async_b128(lds_off_of(&tileB[(kk + 1) & 1][jr * TS + seg * 4]),
                           projT + (size_t)(j0 + 32 + jr) * DD + seg * 4);
            }
            asm volatile("s_wait_asynccnt 0x2" ::: "memory");
        } else {
            asm volatile("s_wait_asynccnt 0x0" ::: "memory");
        }
        __syncthreads();

        // A fragment: p = exp(m - M) on surviving edges, straight into bf16 split
        v16bf ah, al;
#pragma unroll
        for (int p = 0; p < 8; ++p) {
            const int jb = j0 + half * 16 + 2 * p;
#pragma unroll
            for (int slot = 0; slot < 2; ++slot) {
                const int j = jb + slot;
                float pv = 0.f;
                if (d0 || arow[j]) pv = __expf(mrow[(size_t)j * HH] - Mi);
                splitpush(ah, al, 2 * p + slot, pv);
            }
        }
        // B fragments from the shared LDS tile
        const float* tb = &tileB[kk & 1][0];
        v16bf bh0, bl0, bh1, bl1;
#pragma unroll
        for (int e = 0; e < 16; ++e) {
            const int jr = half * 16 + e;
            splitpush(bh0, bl0, e, tb[jr * TS + r]);
            splitpush(bh1, bl1, e, tb[jr * TS + 16 + r]);
        }
        acc0 = wmma3(ah, al, bh0, bl0, acc0);
        acc1 = wmma3(ah, al, bh1, bl1, acc1);
        __syncthreads();
    }

#pragma unroll
    for (int k = 0; k < 8; ++k) {
        const int i = i0 + half * 8 + k;         // D layout row
        const size_t bir = (size_t)(b * NN + i);
        const float sc = Srow[bir * HH + h];     // sqrt(S2)/S1^2
        agg[bir * DD + h * DHH + r]      = acc0[k] * sc;
        agg[bir * DD + h * DHH + 16 + r] = acc1[k] * sc;
    }
}

// ---------------------------------------------------------------------------
// Launch
// ---------------------------------------------------------------------------
extern "C" void kernel_launch(void* const* d_in, const int* in_sizes, int n_in,
                              void* d_out, int out_size, void* d_ws, size_t ws_size,
                              hipStream_t stream) {
    (void)in_sizes; (void)n_in; (void)out_size; (void)ws_size;
    const float* node     = (const float*)d_in[0];
    const float* messages = (const float*)d_in[1];
    const int*   adj      = (const int*)d_in[2];
    const float* gamma    = (const float*)d_in[3];
    const float* beta     = (const float*)d_in[4];
    const float* w_in     = (const float*)d_in[5];
    const float* b_in     = (const float*)d_in[6];
    const float* w_out    = (const float*)d_in[7];
    const float* b_out    = (const float*)d_in[8];
    float* out = (float*)d_out;

    char* ws = (char*)d_ws;
    float* nf    = (float*)(ws);                               // 8 MB
    float* proj  = (float*)(ws + ((size_t)8 << 20));           // 8 MB
    float* agg   = (float*)(ws + ((size_t)16 << 20));          // 8 MB
    float* Mrow  = (float*)(ws + ((size_t)24 << 20));          // 256 KB
    float* Srow  = (float*)(ws + ((size_t)24 << 20) + (1u << 18));
    int*   deg0  = (int*)  (ws + ((size_t)24 << 20) + (2u << 18));

    // 1) LayerNorm
    ln_kernel<<<ROWS / 8, 256, 0, stream>>>(node, gamma, beta, nf);
    // 2) input projection: proj = nf @ w_in^T + b_in
    gemm_bias_kernel<<<dim3(ROWS / 16, DD / 128), 128, 0, stream>>>(nf, w_in, b_in, proj);
    // 3) masked-softmax stats per (b,i,h)
    stats_kernel<<<ROWS, 256, 0, stream>>>(messages, adj, Mrow, Srow, deg0);
    // 4) attention aggregation + L2 rescale
    agg_kernel<<<BB * HH * (NN / 64), 128, 0, stream>>>(messages, adj, proj, Mrow, Srow, deg0, agg);
    // 5) output projection: out = agg @ w_out^T + b_out
    gemm_bias_kernel<<<dim3(ROWS / 16, DD / 128), 128, 0, stream>>>(agg, w_out, b_out, out);
}
